// AttentionLayer_10333691314616
// MI455X (gfx1250) — matrix operand order repaired and checked by reading, not compiled, not run on hardware
//
#include <hip/hip_runtime.h>
#include <hip/hip_bf16.h>
#include <math.h>

// ---------------------------------------------------------------------------
// AttentionLayer for MI455X (gfx1250, wave32, WMMA).
// Flash-attention with v_wmma_f32_16x16x32_f16 for QK^T and P*V.
// Round 2: DPP16 butterfly reductions (no ds_bpermute), base-2 softmax with
// log2(e) folded into the Q scale, register-prefetch pipelining of K/V.
// ---------------------------------------------------------------------------

typedef __attribute__((ext_vector_type(16))) _Float16 v16h;
typedef __attribute__((ext_vector_type(8)))  float    v8f;

#define BATCH 8
#define S_LEN 4096
#define HDIM  64
#define NPB   262144   // S_LEN*HDIM elements per batch

// --- DPP16 cross-lane helpers (16-lane butterfly, stays inside each half) ---
template <int CTRL>
__device__ __forceinline__ float dpp_movf(float x) {
  return __int_as_float(
      __builtin_amdgcn_update_dpp(0, __float_as_int(x), CTRL, 0xf, 0xf, true));
}
// quad_perm(1,0,3,2)=0xB1 (xor1), quad_perm(2,3,0,1)=0x4E (xor2),
// row_ror:4=0x124, row_ror:8=0x128 — together a full 16-lane all-reduce.
__device__ __forceinline__ float rowmax16(float x) {
  x = fmaxf(x, dpp_movf<0xB1>(x));
  x = fmaxf(x, dpp_movf<0x4E>(x));
  x = fmaxf(x, dpp_movf<0x124>(x));
  x = fmaxf(x, dpp_movf<0x128>(x));
  return x;
}
__device__ __forceinline__ float rowsum16(float x) {
  x += dpp_movf<0xB1>(x);
  x += dpp_movf<0x4E>(x);
  x += dpp_movf<0x124>(x);
  x += dpp_movf<0x128>(x);
  return x;
}

// ---------------------------------------------------------------------------
// Stage 1: q/k/v projections (in_dim=2 -> 64). Tiny; memory-trivial.
//   qf fp32 [b][s][h]   (residual + A-matrix source)
//   kh f16  [b][key][h] (row-major: score B-matrix reads h-contiguous)
//   vt f16  [b][h][key] (transposed: P*V B-matrix reads key-contiguous)
// ---------------------------------------------------------------------------
__global__ __launch_bounds__(256) void qkv_kernel(
    const float* __restrict__ x,
    const float* __restrict__ Wq, const float* __restrict__ bq,
    const float* __restrict__ Wk, const float* __restrict__ bk,
    const float* __restrict__ Wv, const float* __restrict__ bv,
    float* __restrict__ qf, _Float16* __restrict__ kh, _Float16* __restrict__ vt)
{
  int idx = blockIdx.x * 256 + threadIdx.x;      // [0, B*S*H)
  int h  = idx & (HDIM - 1);
  int bs = idx >> 6;                             // b*S + s
  int s  = bs & (S_LEN - 1);
  int b  = bs >> 12;
  float x0 = x[bs * 2 + 0], x1 = x[bs * 2 + 1];
  float q = x0 * Wq[h * 2] + x1 * Wq[h * 2 + 1] + bq[h];
  float k = x0 * Wk[h * 2] + x1 * Wk[h * 2 + 1] + bk[h];
  float v = x0 * Wv[h * 2] + x1 * Wv[h * 2 + 1] + bv[h];
  qf[idx] = q;
  kh[idx] = (_Float16)k;
  vt[((size_t)(b * HDIM + h)) * S_LEN + s] = (_Float16)v;
}

// ---------------------------------------------------------------------------
// Stage 2: flash attention. 8 waves/block, 16 q-rows per wave (128 rows/block).
// grid = B * (S/128) = 256 blocks. Key loop in 32-key blocks staged in LDS.
// Softmax runs in base-2: Q is pre-scaled by (1/sqrt(H)) * log2(e).
// ---------------------------------------------------------------------------
__global__ __launch_bounds__(256) void flash_kernel(
    const float* __restrict__ qf, const _Float16* __restrict__ kh,
    const _Float16* __restrict__ vt, float* __restrict__ outw)
{
  __shared__ __align__(16) _Float16 k_lds[32 * 72];       // [key][h], stride 72
  __shared__ __align__(16) _Float16 vt_lds[64 * 40];      // [h][key], stride 40
  __shared__ __align__(16) _Float16 p_lds[8][16 * 40];    // per-wave P tile

  const int tid  = threadIdx.x;
  const int wave = tid >> 5;
  const int lane = tid & 31;
  const int hi16 = lane >> 4;    // 16-lane half (selects row/K sub-range)
  const int l16  = lane & 15;

  const int b    = blockIdx.x >> 5;          // 32 q-blocks per batch
  const int q0   = (blockIdx.x & 31) * 128 + wave * 16;

  // --- Q tile in A-matrix layout (ISA 7.12.2), scale = 0.125 * log2(e).
  // A[m][k]: m = lane%16 ; half j -> k = j + 16*(lane/16) (contiguous, same as B), per 32-chunk w.
  const float QSCALE = 0.125f * 1.4426950408889634f;
  v16h Aq[2];
  {
    const float* qrow = qf + ((size_t)(b * S_LEN + q0 + l16)) * HDIM;
#pragma unroll
    for (int w = 0; w < 2; ++w)
#pragma unroll
      for (int j = 0; j < 16; ++j) {
        int h = 32 * w + j + 16 * hi16;
        Aq[w][j] = (_Float16)(qrow[h] * QSCALE);
      }
  }

  v8f Oacc[4] = {v8f{}, v8f{}, v8f{}, v8f{}};
  float mrow[8], lrow[8];
#pragma unroll
  for (int i = 0; i < 8; ++i) { mrow[i] = -INFINITY; lrow[i] = 0.f; }

  const _Float16* kbase = kh + (size_t)b * S_LEN * HDIM;
  const _Float16* vbase = vt + (size_t)b * HDIM * S_LEN;

  // register prefetch of the first K/V block
  const int kr = tid >> 3, kc = tid & 7;   // K: 32 rows x 8 x 16B
  const int vr = tid >> 2, vc = tid & 3;   // V^T: 64 rows x 4 x 16B
  uint4 kreg = *((const uint4*)(kbase + (size_t)kr * HDIM) + kc);
  uint4 vreg = *((const uint4*)(vbase + (size_t)vr * S_LEN) + vc);

  for (int kb0 = 0; kb0 < S_LEN; kb0 += 32) {
    // commit prefetched block to LDS
    *((uint4*)(k_lds + kr * 72) + kc) = kreg;
    *((uint4*)(vt_lds + vr * 40) + vc) = vreg;
    __syncthreads();
    // issue next block's global loads early (overlap with compute below)
    if (kb0 + 32 < S_LEN) {
      kreg = *((const uint4*)(kbase + (size_t)(kb0 + 32 + kr) * HDIM) + kc);
      vreg = *((const uint4*)(vbase + (size_t)vr * S_LEN + (kb0 + 32)) + vc);
    }

    // scores for two 16-key sub-tiles: S2 = (Q*scale*log2e) @ K^T
    v8f st[2];
#pragma unroll
    for (int t16 = 0; t16 < 2; ++t16) {
      v8f acc = {};
#pragma unroll
      for (int w = 0; w < 2; ++w) {
        // B[k][n]: n = key = lane%16 ; half j -> k = j + 16*(lane/16); h = 32w + k
        v16h Bk;
        const _Float16* krow = k_lds + (t16 * 16 + l16) * 72 + 32 * w + 16 * hi16;
#pragma unroll
        for (int j = 0; j < 16; ++j) Bk[j] = krow[j];
        acc = __builtin_amdgcn_wmma_f32_16x16x32_f16(
            false, Aq[w], false, Bk, (short)0, acc, false, false);
      }
      st[t16] = acc;
    }

    // --- online softmax (base 2) across the 32-key block; row = slot i + 8*hi16
    float nm[8], alpha[8];
#pragma unroll
    for (int i = 0; i < 8; ++i) {
      float r = rowmax16(fmaxf(st[0][i], st[1][i]));
      float mn = fmaxf(mrow[i], r);
      nm[i]    = mn;
      alpha[i] = exp2f(mrow[i] - mn);   // 0 on first block (exp2(-inf))
      mrow[i]  = mn;
    }
#pragma unroll
    for (int i = 0; i < 8; ++i) {
      float p0 = exp2f(st[0][i] - nm[i]);
      float p1 = exp2f(st[1][i] - nm[i]);
      st[0][i] = p0; st[1][i] = p1;
      float s = rowsum16(p0 + p1);
      lrow[i] = lrow[i] * alpha[i] + s;
    }

    // --- C-layout -> A-layout transpose of P via per-wave LDS bounce
    _Float16* pw = p_lds[wave];
#pragma unroll
    for (int t16 = 0; t16 < 2; ++t16)
#pragma unroll
      for (int i = 0; i < 8; ++i)
        pw[(i + 8 * hi16) * 40 + l16 + 16 * t16] = (_Float16)st[t16][i];
    v16h Ap;
    {
      const _Float16* pr = pw + l16 * 40;
#pragma unroll
      for (int j = 0; j < 16; ++j)
        Ap[j] = pr[j + 16 * hi16];
    }

    // --- rescale running O by alpha, then O += P @ V (4 N-chunks of H)
#pragma unroll
    for (int c = 0; c < 4; ++c)
#pragma unroll
      for (int i = 0; i < 8; ++i) Oacc[c][i] *= alpha[i];
#pragma unroll
    for (int c = 0; c < 4; ++c) {
      // B[k][n]: k = key = j + 16*(lane/16); n = h = lane%16 + 16*c  (from V^T)
      v16h Bv;
      const _Float16* vrow = vt_lds + (l16 + 16 * c) * 40 + 16 * hi16;
#pragma unroll
      for (int j = 0; j < 16; ++j) Bv[j] = vrow[j];
      Oacc[c] = __builtin_amdgcn_wmma_f32_16x16x32_f16(
          false, Ap, false, Bv, (short)0, Oacc[c], false, false);
    }
    __syncthreads();   // protect LDS before next stage
  }

  // epilogue: normalize rows by 1/l and store fp32 "weighted"
  float inv[8];
#pragma unroll
  for (int i = 0; i < 8; ++i) inv[i] = 1.0f / lrow[i];
#pragma unroll
  for (int c = 0; c < 4; ++c)
#pragma unroll
    for (int i = 0; i < 8; ++i)
      outw[((size_t)(b * S_LEN + q0 + i + 8 * hi16)) * HDIM + 16 * c + l16] =
          Oacc[c][i] * inv[i];
}

// ---------------------------------------------------------------------------
// Stage 3a: t = weighted + q (in place) + per-block partial sums for LN1.
// grid = B*128 blocks; 2048 elements/block (8/thread).
// ---------------------------------------------------------------------------
__global__ __launch_bounds__(256) void addres_partial(
    const float* __restrict__ qf, float* __restrict__ t,
    float* __restrict__ ps, float* __restrict__ pq)
{
  __shared__ float r1[256], r2[256];
  int bchunk = blockIdx.x;                    // b*128 + chunk
  size_t base = (size_t)(bchunk >> 7) * NPB + (size_t)(bchunk & 127) * 2048
              + (size_t)threadIdx.x * 8;
  float s = 0.f, sq = 0.f;
#pragma unroll
  for (int i = 0; i < 8; ++i) {
    float v = t[base + i] + qf[base + i];
    t[base + i] = v;
    s += v; sq += v * v;
  }
  r1[threadIdx.x] = s; r2[threadIdx.x] = sq;
  __syncthreads();
  for (int off = 128; off > 0; off >>= 1) {
    if (threadIdx.x < off) {
      r1[threadIdx.x] += r1[threadIdx.x + off];
      r2[threadIdx.x] += r2[threadIdx.x + off];
    }
    __syncthreads();
  }
  if (threadIdx.x == 0) { ps[blockIdx.x] = r1[0]; pq[blockIdx.x] = r2[0]; }
}

// ---------------------------------------------------------------------------
// Reduce partials -> per-batch {mu, rsqrt(var+eps)}. grid = B.
// ---------------------------------------------------------------------------
__global__ __launch_bounds__(256) void reduce_stats(
    const float* __restrict__ ps, const float* __restrict__ pq,
    int nch, float invN, float* __restrict__ stats)
{
  __shared__ float r1[256], r2[256];
  int b = blockIdx.x;
  float s = 0.f, q = 0.f;
  for (int i = threadIdx.x; i < nch; i += 256) {
    s += ps[b * nch + i]; q += pq[b * nch + i];
  }
  r1[threadIdx.x] = s; r2[threadIdx.x] = q;
  __syncthreads();
  for (int off = 128; off > 0; off >>= 1) {
    if (threadIdx.x < off) {
      r1[threadIdx.x] += r1[threadIdx.x + off];
      r2[threadIdx.x] += r2[threadIdx.x + off];
    }
    __syncthreads();
  }
  if (threadIdx.x == 0) {
    float mu  = r1[0] * invN;
    float var = r2[0] * invN - mu * mu;
    stats[b * 2 + 0] = mu;
    stats[b * 2 + 1] = rsqrtf(var + 1e-5f);
  }
}

// ---------------------------------------------------------------------------
// Stage 3c: h = LN1(t); ff = relu(h@W0^T+b0); out_pre = h + ff@W1^T + b1.
// 64 threads per row, 4 rows per block. grid = B*1024. Also emits LN2 partials.
// ---------------------------------------------------------------------------
__global__ __launch_bounds__(256) void ffn_kernel(
    const float* __restrict__ t, const float* __restrict__ stats,
    const float* __restrict__ W0, const float* __restrict__ b0,
    const float* __restrict__ W1, const float* __restrict__ b1,
    float* __restrict__ outp, float* __restrict__ ps2, float* __restrict__ pq2)
{
  __shared__ float hrow[4][64];
  __shared__ float ffl[4][128];
  __shared__ float r1[256], r2[256];
  int b    = blockIdx.x >> 10;
  int rb   = blockIdx.x & 1023;
  int grp  = threadIdx.x >> 6;
  int l    = threadIdx.x & 63;
  size_t rbase = ((size_t)b * S_LEN + rb * 4 + grp) * HDIM;
  float mu = stats[b * 2], rs = stats[b * 2 + 1];
  hrow[grp][l] = (t[rbase + l] - mu) * rs;
  __syncthreads();
  for (int f = l; f < 128; f += 64) {
    float a = b0[f];
#pragma unroll 8
    for (int i = 0; i < 64; ++i) a += hrow[grp][i] * W0[f * 64 + i];
    ffl[grp][f] = fmaxf(a, 0.f);
  }
  __syncthreads();
  float a = hrow[grp][l] + b1[l];
#pragma unroll 8
  for (int f = 0; f < 128; ++f) a += ffl[grp][f] * W1[l * 128 + f];
  outp[rbase + l] = a;
  r1[threadIdx.x] = a; r2[threadIdx.x] = a * a;
  __syncthreads();
  for (int off = 128; off > 0; off >>= 1) {
    if (threadIdx.x < off) {
      r1[threadIdx.x] += r1[threadIdx.x + off];
      r2[threadIdx.x] += r2[threadIdx.x + off];
    }
    __syncthreads();
  }
  if (threadIdx.x == 0) { ps2[blockIdx.x] = r1[0]; pq2[blockIdx.x] = r2[0]; }
}

// ---------------------------------------------------------------------------
// Stage 3e: final layernorm -> d_out.
// ---------------------------------------------------------------------------
__global__ __launch_bounds__(256) void norm_kernel(
    const float* __restrict__ outp, const float* __restrict__ stats,
    float* __restrict__ out)
{
  size_t idx = (size_t)blockIdx.x * 256 + threadIdx.x;
  int b = (int)(idx >> 18);   // 2^18 elements per batch
  out[idx] = (outp[idx] - stats[b * 2]) * stats[b * 2 + 1];
}

// ---------------------------------------------------------------------------
extern "C" void kernel_launch(void* const* d_in, const int* in_sizes, int n_in,
                              void* d_out, int out_size, void* d_ws, size_t ws_size,
                              hipStream_t stream)
{
  const float* x  = (const float*)d_in[0];
  const float* Wq = (const float*)d_in[1];
  const float* bq = (const float*)d_in[2];
  const float* Wk = (const float*)d_in[3];
  const float* bk = (const float*)d_in[4];
  const float* Wv = (const float*)d_in[5];
  const float* bv = (const float*)d_in[6];
  const float* W0 = (const float*)d_in[7];
  const float* b0 = (const float*)d_in[8];
  const float* W1 = (const float*)d_in[9];
  const float* b1 = (const float*)d_in[10];
  float* out = (float*)d_out;

  char* w = (char*)d_ws;
  const size_t NE = (size_t)BATCH * S_LEN * HDIM;     // 2M elements
  float*    qf   = (float*)(w);                       //  8 MB
  _Float16* kh   = (_Float16*)(w + NE * 4);           //  4 MB
  _Float16* vt   = (_Float16*)(w + NE * 6);           //  4 MB
  float*    wt   = (float*)(w + NE * 8);              //  8 MB (weighted -> t)
  float*    outp = (float*)(w + NE * 12);             //  8 MB
  float*    ps1  = (float*)(w + NE * 16);             // 1024
  float*    pq1  = ps1 + 1024;
  float*    ps2  = pq1 + 1024;                        // 8192
  float*    pq2  = ps2 + 8192;
  float*    st1  = pq2 + 8192;                        // 16
  float*    st2  = st1 + 16;

  qkv_kernel<<<8192, 256, 0, stream>>>(x, Wq, bq, Wk, bk, Wv, bv, qf, kh, vt);
  flash_kernel<<<BATCH * (S_LEN / 128), 256, 0, stream>>>(qf, kh, vt, wt);
  addres_partial<<<BATCH * 128, 256, 0, stream>>>(qf, wt, ps1, pq1);
  reduce_stats<<<BATCH, 256, 0, stream>>>(ps1, pq1, 128, 1.0f / (float)NPB, st1);
  ffn_kernel<<<BATCH * 1024, 256, 0, stream>>>(wt, st1, W0, b0, W1, b1, outp, ps2, pq2);
  reduce_stats<<<BATCH, 256, 0, stream>>>(ps2, pq2, 1024, 1.0f / (float)NPB, st2);
  norm_kernel<<<8192, 256, 0, stream>>>(outp, st2, out);
}
